// GraphAttentionLayer_8589935196
// MI455X (gfx1250) — matrix operand order repaired and checked by reading, not compiled, not run on hardware
//
#include <hip/hip_runtime.h>
#include <hip/hip_bf16.h>

typedef __bf16 bf16_t;
typedef __attribute__((ext_vector_type(16))) __bf16 v16bf;
typedef __attribute__((ext_vector_type(8)))  float  v8f;
typedef __attribute__((ext_vector_type(4)))  int    v4i;

union AFrag { v16bf v; bf16_t h[16]; uint4 q[2]; };
union CFrag { v8f v; float f[8]; };

#define BATCH  8
#define NN     2048
#define INDIM  128
#define OUTDIM 128
#define HEADS  4
#define HDIM   32
#define SWP    136      // padded LDS row stride for W^T (bank de-conflict)
#define SVP    2056     // padded LDS row stride for V slice (bank de-conflict)

#define AS1 __attribute__((address_space(1)))
#define AS3 __attribute__((address_space(3)))

#if defined(__has_builtin)
#  if __has_builtin(__builtin_amdgcn_global_load_async_to_lds_b128)
#    define HAVE_ASYNC_LDS 1
#  endif
#  if __has_builtin(__builtin_amdgcn_s_wait_asynccnt)
#    define HAVE_WAIT_ASYNC 1
#  endif
#endif

// ---------------------------------------------------------------------------
// Kernel 1: Wh = h @ W   (16384x128 @ 128x128), bf16 WMMA.
// W is staged into LDS *transposed* (sWT[n][k], padded stride) so each
// B fragment (two runs of 8 K values per lane, 16 apart) is 2 x ds_load_b128.
// Block = 256 threads (8 waves); block owns a 16-row M tile, wave owns a
// 16-col N tile; K=128 -> 4 x v_wmma_f32_16x16x32_bf16 per wave.
// ---------------------------------------------------------------------------
__global__ __launch_bounds__(256)
void gat_gemm_wh(const float* __restrict__ h, const float* __restrict__ W,
                 float* __restrict__ Wh)
{
    __shared__ bf16_t sWT[OUTDIM * SWP];     // ~34 KB

    const int t = threadIdx.x;
    {   // cooperative stage: W row-major f32 -> transposed bf16 in LDS
        const float4* W4 = (const float4*)W;
        const int base = t * 16;             // 16 float4 per thread
        #pragma unroll
        for (int q = 0; q < 16; ++q) {
            float4 v = W4[base + q];
            int el = (base + q) * 4;         // element = k*128 + n
            int k  = el >> 7;
            int n  = el & 127;
            sWT[(n + 0) * SWP + k] = (bf16_t)v.x;
            sWT[(n + 1) * SWP + k] = (bf16_t)v.y;
            sWT[(n + 2) * SWP + k] = (bf16_t)v.z;
            sWT[(n + 3) * SWP + k] = (bf16_t)v.w;
        }
    }
    __syncthreads();

    const int lane = t & 31;
    const int wid  = t >> 5;                 // 0..7 -> N tile
    const int hi   = lane >> 4;
    const int lo   = lane & 15;
    const int row  = blockIdx.x * 16 + lo;   // global M row (B*N flattened)
    const int ncol = wid * 16 + lo;          // global N col

    CFrag c;
    #pragma unroll
    for (int i = 0; i < 8; ++i) c.f[i] = 0.f;

    const float* hrow = h + (size_t)row * INDIM;

    #pragma unroll
    for (int kb = 0; kb < INDIM; kb += 32) {
        // A fragment (16x32 bf16): lane half picks K runs {0..7,16..23}/{8..15,24..31}
        AFrag a;
        float4 r0 = *(const float4*)(hrow + kb + 8 * hi);
        float4 r1 = *(const float4*)(hrow + kb + 8 * hi + 4);
        float4 r2 = *(const float4*)(hrow + kb + 16 + 8 * hi);
        float4 r3 = *(const float4*)(hrow + kb + 16 + 8 * hi + 4);
        a.h[ 0] = (bf16_t)r0.x; a.h[ 1] = (bf16_t)r0.y; a.h[ 2] = (bf16_t)r0.z; a.h[ 3] = (bf16_t)r0.w;
        a.h[ 4] = (bf16_t)r1.x; a.h[ 5] = (bf16_t)r1.y; a.h[ 6] = (bf16_t)r1.z; a.h[ 7] = (bf16_t)r1.w;
        a.h[ 8] = (bf16_t)r2.x; a.h[ 9] = (bf16_t)r2.y; a.h[10] = (bf16_t)r2.z; a.h[11] = (bf16_t)r2.w;
        a.h[12] = (bf16_t)r3.x; a.h[13] = (bf16_t)r3.y; a.h[14] = (bf16_t)r3.z; a.h[15] = (bf16_t)r3.w;

        // B fragment: lane half picks K runs {0..7,16..23}/{8..15,24..31}
        AFrag bfr;
        const bf16_t* wp = &sWT[ncol * SWP + kb + 8 * hi];
        bfr.q[0] = *(const uint4*)(wp);
        bfr.q[1] = *(const uint4*)(wp + 16);

        c.v = __builtin_amdgcn_wmma_f32_16x16x32_bf16(
                  false, a.v, false, bfr.v, (short)0, c.v, false, false);
    }

    #pragma unroll
    for (int r = 0; r < 8; ++r) {
        int m = blockIdx.x * 16 + r + 8 * hi;    // C layout: M = r + 8*laneHalf
        Wh[(size_t)m * OUTDIM + ncol] = c.f[r];
    }
}

// ---------------------------------------------------------------------------
// Kernel 2a: attention logit scalars Wh1 = Wh.a1, Wh2 = Wh.a2 per (h,b,n).
// ---------------------------------------------------------------------------
__global__ __launch_bounds__(256)
void gat_prep(const float* __restrict__ Wh, const float* __restrict__ a,
              float* __restrict__ Wh1, float* __restrict__ Wh2)
{
    const int tid = blockIdx.x * blockDim.x + threadIdx.x;  // H*B*N = 65536
    const int n  = tid & (NN - 1);
    const int b  = (tid >> 11) & (BATCH - 1);
    const int hh = tid >> 14;

    const float* row = Wh + (size_t)(b * NN + n) * OUTDIM + hh * HDIM;
    float s1 = 0.f, s2 = 0.f;
    #pragma unroll
    for (int d = 0; d < HDIM; d += 4) {
        float4 v = *(const float4*)(row + d);
        s1 += v.x * a[d]   + v.y * a[d+1]   + v.z * a[d+2]   + v.w * a[d+3];
        s2 += v.x * a[HDIM+d] + v.y * a[HDIM+d+1] + v.z * a[HDIM+d+2] + v.w * a[HDIM+d+3];
    }
    Wh1[(hh * BATCH + b) * NN + n] = s1;
    Wh2[(hh * BATCH + b) * NN + n] = s2;
}

// ---------------------------------------------------------------------------
// Kernel 2b: LDS-tiled transpose Wh (B,N,128) f32 -> WhT (H,B,D,N) bf16.
// Coalesced float4 reads, padded LDS tile, coalesced uint4 (b128) writes.
// Block handles 128 j-rows x 32 d for one (h,b). 32 slices * 16 tiles = 512.
// ---------------------------------------------------------------------------
__global__ __launch_bounds__(256)
void gat_transpose(const float* __restrict__ Wh, bf16_t* __restrict__ WhT)
{
    __shared__ bf16_t sT[HDIM * SWP];        // 32 x 136

    const int t = threadIdx.x;
    const int slice = blockIdx.x >> 4;       // hh*BATCH + b
    const int b  = slice & (BATCH - 1);
    const int hh = slice >> 3;
    const int j0 = (blockIdx.x & 15) * 128;

    #pragma unroll
    for (int p = 0; p < 4; ++p) {
        int idx = t + p * 256;               // 1024 float4 = 128 rows x 8
        int jl  = idx >> 3;
        int d4  = idx & 7;
        float4 v = *(const float4*)(Wh + (size_t)(b * NN + j0 + jl) * OUTDIM
                                       + hh * HDIM + d4 * 4);
        int d = d4 * 4;
        sT[(d + 0) * SWP + jl] = (bf16_t)v.x;
        sT[(d + 1) * SWP + jl] = (bf16_t)v.y;
        sT[(d + 2) * SWP + jl] = (bf16_t)v.z;
        sT[(d + 3) * SWP + jl] = (bf16_t)v.w;
    }
    __syncthreads();

    bf16_t* dst = WhT + (size_t)slice * HDIM * NN;
    #pragma unroll
    for (int q = 0; q < 2; ++q) {
        int idx = t + q * 256;               // 512 uint4 = 32 rows x 16
        int d   = idx >> 4;
        int jq  = idx & 15;
        uint4 v = *(const uint4*)&sT[d * SWP + jq * 8];
        *(uint4*)(dst + (size_t)d * NN + j0 + jq * 8) = v;
    }
}

// ---------------------------------------------------------------------------
// Kernel 3: fused masked-softmax attention, flash style, single pass over A.
// Block = 256 threads (8 waves), all waves share one (head,batch): the whole
// V slice (32 x 2048 bf16 = 128 KB) is staged into LDS once per block via
// gfx1250 async global->LDS b128 copies (ASYNCcnt), then each wave runs
// online softmax over its 16-row tile with two bf16 WMMAs per 32-j chunk.
// ---------------------------------------------------------------------------
__global__ __launch_bounds__(256)
void gat_attn(const int* __restrict__ A, const float* __restrict__ Wh1,
              const float* __restrict__ Wh2, const bf16_t* __restrict__ WhT,
              float* __restrict__ out)
{
    extern __shared__ char smem[];
    bf16_t* sV = (bf16_t*)smem;              // [HDIM][SVP] padded, ~128.5 KB

    const int lane = threadIdx.x & 31;
    const int wid  = threadIdx.x >> 5;       // 0..7
    const int tile = blockIdx.x * 8 + wid;   // 4096 tiles total
    const int it = tile & 127;               // row tile within (h,b)
    const int b  = (tile >> 7) & (BATCH - 1);
    const int hh = tile >> 10;
    const int hi = lane >> 4;
    const int lo = lane & 15;
    const int row = it * 16 + lo;

    // ---- stage V slice (shared by all 8 waves: tile>>7 is block-uniform) ----
    const bf16_t* gsl = WhT + (size_t)(hh * BATCH + b) * HDIM * NN;
    #pragma unroll
    for (int i = 0; i < 32; ++i) {
        int idx = threadIdx.x + i * 256;     // 8192 x 16B chunks
        int d = idx >> 8;
        int j = (idx & 255) * 8;
#if defined(HAVE_ASYNC_LDS)
        __builtin_amdgcn_global_load_async_to_lds_b128(
            (AS1 v4i*)(AS1 void*)(void*)(gsl + (size_t)d * NN + j),
            (AS3 v4i*)(AS3 void*)(void*)&sV[d * SVP + j], 0, 0);
#else
        *(uint4*)&sV[d * SVP + j] = *(const uint4*)(gsl + (size_t)d * NN + j);
#endif
    }
#if defined(HAVE_ASYNC_LDS)
#  if defined(HAVE_WAIT_ASYNC)
    __builtin_amdgcn_s_wait_asynccnt(0);
#  else
    asm volatile("s_wait_asynccnt 0x0" ::: "memory");
#  endif
#endif
    __syncthreads();

    const float  wh1  = Wh1[(hh * BATCH + b) * NN + row];
    const int*   Arow = A + (size_t)b * NN * NN + (size_t)row * NN;
    const float* w2p  = Wh2 + (hh * BATCH + b) * NN;

    float m = -1e30f, l = 0.f;
    CFrag acc0, acc1;
    #pragma unroll
    for (int i = 0; i < 8; ++i) { acc0.f[i] = 0.f; acc1.f[i] = 0.f; }

    for (int jc = 0; jc < NN; jc += 32) {
        __builtin_prefetch(Arow + jc + 128, 0, 0);   // stream-ahead on A (HBM roofline term)

        // K runs matching the 16-bit A-fragment layout for this lane half
        const int k0 = jc + 8 * hi;
        const int k1 = jc + 16 + 8 * hi;
        int4 ma = *(const int4*)(Arow + k0);
        int4 mb = *(const int4*)(Arow + k0 + 4);
        int4 mc = *(const int4*)(Arow + k1);
        int4 md = *(const int4*)(Arow + k1 + 4);
        float4 wa = *(const float4*)(w2p + k0);
        float4 wb = *(const float4*)(w2p + k0 + 4);
        float4 wc = *(const float4*)(w2p + k1);
        float4 wd = *(const float4*)(w2p + k1 + 4);

        const float w2v[16] = { wa.x, wa.y, wa.z, wa.w,  wb.x, wb.y, wb.z, wb.w,
                                wc.x, wc.y, wc.z, wc.w,  wd.x, wd.y, wd.z, wd.w };
        const int   mk[16]  = { ma.x, ma.y, ma.z, ma.w,  mb.x, mb.y, mb.z, mb.w,
                                mc.x, mc.y, mc.z, mc.w,  md.x, md.y, md.z, md.w };

        float e[16];
        #pragma unroll
        for (int i = 0; i < 16; ++i) {
            float ev = wh1 + w2v[i];
            ev = ev > 0.f ? ev : 0.2f * ev;          // leaky_relu(0.2)
            e[i] = mk[i] > 0 ? ev : -1e30f;          // adjacency mask
        }

        float cm = e[0];
        #pragma unroll
        for (int i = 1; i < 16; ++i) cm = fmaxf(cm, e[i]);
        cm = fmaxf(cm, __shfl_xor(cm, 16, 32));      // lane pair covers all 32 j
        const float mnew  = fmaxf(m, cm);
        const float scale = __expf(m - mnew);
        m = mnew;

        AFrag p;
        float psum = 0.f;
        #pragma unroll
        for (int i = 0; i < 16; ++i) {
            float pv = (e[i] < -1e29f) ? 0.f : __expf(e[i] - mnew);
            psum += pv;
            p.h[i] = (bf16_t)pv;
        }
        psum += __shfl_xor(psum, 16, 32);
        l = l * scale + psum;

        // rescale accumulators: C-tile row M = r + 8*hi; scale lives in lane M
        #pragma unroll
        for (int r = 0; r < 8; ++r) {
            float s = __shfl(scale, r + 8 * hi, 32);
            acc0.f[r] *= s;
            acc1.f[r] *= s;
        }

        // B fragments from LDS V slice: 2 x ds_load_b128 each
        AFrag b0, b1;
        const bf16_t* c0 = &sV[(size_t)lo * SVP + jc + 8 * hi];
        const bf16_t* c1 = &sV[(size_t)(lo + 16) * SVP + jc + 8 * hi];
        b0.q[0] = *(const uint4*)(c0);
        b0.q[1] = *(const uint4*)(c0 + 16);
        b1.q[0] = *(const uint4*)(c1);
        b1.q[1] = *(const uint4*)(c1 + 16);

        acc0.v = __builtin_amdgcn_wmma_f32_16x16x32_bf16(
                     false, p.v, false, b0.v, (short)0, acc0.v, false, false);
        acc1.v = __builtin_amdgcn_wmma_f32_16x16x32_bf16(
                     false, p.v, false, b1.v, (short)0, acc1.v, false, false);
    }

    #pragma unroll
    for (int r = 0; r < 8; ++r) {
        float li  = __shfl(l, r + 8 * hi, 32);
        float inv = 1.0f / li;
        int mrow = it * 16 + r + 8 * hi;
        float* orow = out + (size_t)(b * NN + mrow) * OUTDIM + hh * HDIM;
        orow[lo]      = acc0.f[r] * inv;
        orow[lo + 16] = acc1.f[r] * inv;
    }
}

// ---------------------------------------------------------------------------
extern "C" void kernel_launch(void* const* d_in, const int* in_sizes, int n_in,
                              void* d_out, int out_size, void* d_ws, size_t ws_size,
                              hipStream_t stream)
{
    const float* h = (const float*)d_in[0];   // (B,N,128) f32
    const int*   A = (const int*)  d_in[1];   // (B,N,N)   i32
    const float* W = (const float*)d_in[2];   // (128,128) f32
    const float* a = (const float*)d_in[3];   // (64,1)    f32
    float* out = (float*)d_out;               // (B,N,128) f32

    char* ws = (char*)d_ws;
    float*  Wh  = (float*)ws;                                   // 8 MB
    bf16_t* WhT = (bf16_t*)(ws + (size_t)8 * 1024 * 1024);      // 4 MB
    float*  Wh1 = (float*)(ws + (size_t)12 * 1024 * 1024);      // 256 KB
    float*  Wh2 = Wh1 + HEADS * BATCH * NN;                     // 256 KB

    hipLaunchKernelGGL(gat_gemm_wh, dim3(BATCH * NN / 16), dim3(256), 0, stream,
                       h, W, Wh);
    hipLaunchKernelGGL(gat_prep, dim3(HEADS * BATCH * NN / 256), dim3(256), 0, stream,
                       Wh, a, Wh1, Wh2);
    hipLaunchKernelGGL(gat_transpose, dim3(HEADS * BATCH * 16), dim3(256), 0, stream,
                       Wh, WhT);
    hipLaunchKernelGGL(gat_attn, dim3(HEADS * BATCH * (NN / 16) / 8), dim3(256),
                       HDIM * SVP * (int)sizeof(bf16_t), stream,
                       A, Wh1, Wh2, WhT, out);
}
